// SelfAttention_13168369729698
// MI455X (gfx1250) — compile-verified, matrix operand order checked by reading, not run on hardware
//
#include <hip/hip_runtime.h>

// Self-attention (B=8, S=2048, H=1024, fp32 I/O) for MI455X / gfx1250.
// Compute-bound (~240 GFLOP vs ~25us HBM traffic @23.3TB/s) -> all matmuls
// use V_WMMA_F32_16X16X32_BF16 (bf16 in, fp32 accumulate).
// GEMM microkernel: 32(M)x64(N) per wave, K step 32, software-pipelined:
// fragments for step k+1 are loaded while step k's 8 WMMAs execute.
// All "B" operands stored K-major; softmax emits bf16 probabilities so the
// PV GEMM's A fragments load directly into WMMA operands (no cvt in loop).

#define B_  8
#define S_  2048
#define H_  1024
#define BS_ (B_ * S_)

typedef __attribute__((ext_vector_type(16))) __bf16 v16bf;
typedef __attribute__((ext_vector_type(8)))  __bf16 v8bf;
typedef __attribute__((ext_vector_type(8)))  float  v8f;

__device__ __forceinline__ v8f wmma_bf16(v16bf a, v16bf b, v8f c) {
  return __builtin_amdgcn_wmma_f32_16x16x32_bf16(
      false, a, false, b, (short)0, c, false, false);
}

// Per-lane fragment base pointer (documented 16-bit 16x32 A layout, wave32):
// lane L: row = L&15, K elements half*8+{0..7} and 16+half*8+{0..7}.
__device__ __forceinline__ const char* frag_base_bf16(const __bf16* base, int ld,
                                                      int row0, int lane) {
  return (const char*)(base + (size_t)(row0 + (lane & 15)) * ld + ((lane >> 4) << 3));
}

// koff = k * 2 bytes; lo chunk at +0, hi chunk at +16 elements (+32B).
__device__ __forceinline__ v16bf frag_bf16(const char* __restrict__ p, int koff) {
  v8bf lo = *(const v8bf*)(p + koff);
  v8bf hi = *(const v8bf*)(p + koff + 32);
  return __builtin_shufflevector(lo, hi, 0,1,2,3,4,5,6,7,8,9,10,11,12,13,14,15);
}

// ---------------- elementwise converts ----------------

__global__ void attn_cvt_x(const float* __restrict__ X, __bf16* __restrict__ Xb) {
  size_t i = (size_t)blockIdx.x * blockDim.x + threadIdx.x;
  if (i < (size_t)BS_ * H_) Xb[i] = (__bf16)X[i];
}

// Wt[z][n][k] = W_z[k][n] via LDS tile: coalesced read AND write.
__global__ void attn_cvt_w(const float* __restrict__ Wq, const float* __restrict__ Wk,
                           const float* __restrict__ Wv, __bf16* __restrict__ Wt) {
  __shared__ float tile[32][33];
  const float* W = (blockIdx.z == 0) ? Wq : (blockIdx.z == 1) ? Wk : Wv;
  __bf16* O = Wt + (size_t)blockIdx.z * H_ * H_;
  const int n0 = blockIdx.x * 32, k0 = blockIdx.y * 32;
  const int tx = threadIdx.x, ty = threadIdx.y;
#pragma unroll
  for (int j = 0; j < 32; j += 8)
    tile[ty + j][tx] = W[(size_t)(k0 + ty + j) * H_ + n0 + tx];
  __syncthreads();
#pragma unroll
  for (int j = 0; j < 32; j += 8)
    O[(size_t)(n0 + ty + j) * H_ + k0 + tx] = (__bf16)tile[tx][ty + j];
}

// ======== pipelined 32x64 microkernel (bf16 A and B) ========
// acc0/acc1: rows [row0,row0+16)/[row0+16,row0+32); frags rotate one k-step
// ahead of use so WMMAs never wait on the loads issued in the same step.
#define GEMM_LOOP_BF16(Abase, Ald, Bbase, Bld, KDIM)                          \
  const char* pa0 = frag_base_bf16(Abase, Ald, row0,      lane);              \
  const char* pa1 = frag_base_bf16(Abase, Ald, row0 + 16, lane);              \
  const char* pb0 = frag_base_bf16(Bbase, Bld, col0,      lane);              \
  const char* pb1 = frag_base_bf16(Bbase, Bld, col0 + 16, lane);              \
  const char* pb2 = frag_base_bf16(Bbase, Bld, col0 + 32, lane);              \
  const char* pb3 = frag_base_bf16(Bbase, Bld, col0 + 48, lane);              \
  v16bf a0 = frag_bf16(pa0, 0), a1 = frag_bf16(pa1, 0);                       \
  v16bf b0 = frag_bf16(pb0, 0), b1 = frag_bf16(pb1, 0);                       \
  v16bf b2 = frag_bf16(pb2, 0), b3 = frag_bf16(pb3, 0);                       \
  v8f acc0[4], acc1[4];                                                       \
  _Pragma("unroll")                                                           \
  for (int t = 0; t < 4; ++t) {                                               \
    acc0[t] = (v8f){0.f,0.f,0.f,0.f,0.f,0.f,0.f,0.f};                         \
    acc1[t] = (v8f){0.f,0.f,0.f,0.f,0.f,0.f,0.f,0.f};                         \
  }                                                                           \
  for (int k = 0; k < (KDIM); k += 32) {                                      \
    const int ko = ((k + 32 < (KDIM)) ? (k + 32) : 0) * 2;                    \
    v16bf a0n = frag_bf16(pa0, ko);                                           \
    v16bf a1n = frag_bf16(pa1, ko);                                           \
    v16bf b0n = frag_bf16(pb0, ko);                                           \
    v16bf b1n = frag_bf16(pb1, ko);                                           \
    v16bf b2n = frag_bf16(pb2, ko);                                           \
    v16bf b3n = frag_bf16(pb3, ko);                                           \
    acc0[0] = wmma_bf16(a0, b0, acc0[0]);                                     \
    acc1[0] = wmma_bf16(a1, b0, acc1[0]);                                     \
    acc0[1] = wmma_bf16(a0, b1, acc0[1]);                                     \
    acc1[1] = wmma_bf16(a1, b1, acc1[1]);                                     \
    acc0[2] = wmma_bf16(a0, b2, acc0[2]);                                     \
    acc1[2] = wmma_bf16(a1, b2, acc1[2]);                                     \
    acc0[3] = wmma_bf16(a0, b3, acc0[3]);                                     \
    acc1[3] = wmma_bf16(a1, b3, acc1[3]);                                     \
    a0 = a0n; a1 = a1n; b0 = b0n; b1 = b1n; b2 = b2n; b3 = b3n;               \
  }

// ---------------- QKV projection: [BS,H] x [H,H] + bias ----------------
// z=0 -> Q[s][d], z=1 -> K[s][d], z=2 -> Vt[b][d][s] (contiguous v8bf store).

__global__ void __launch_bounds__(128)
attn_qkv(const __bf16* __restrict__ Xb, const __bf16* __restrict__ Wt,
         const float* __restrict__ bq, const float* __restrict__ bk,
         const float* __restrict__ bv,
         __bf16* __restrict__ Qb, __bf16* __restrict__ Kb, __bf16* __restrict__ Vt) {
  const int z    = blockIdx.z;
  const int lane = threadIdx.x & 31;
  const int wid  = threadIdx.x >> 5;
  const int row0 = blockIdx.x * 64 + (wid & 1) * 32;   // global s
  const int col0 = blockIdx.y * 128 + (wid >> 1) * 64; // output d
  const __bf16* W    = Wt + (size_t)z * H_ * H_;
  const float*  bias = (z == 0) ? bq : (z == 1) ? bk : bv;

  GEMM_LOOP_BF16(Xb, H_, W, H_, H_)

  const int n = lane & 15, hf = lane >> 4;
  if (z < 2) {
    __bf16* Out = (z == 0) ? Qb : Kb;
#pragma unroll
    for (int t = 0; t < 4; ++t) {
      const int   col = col0 + t * 16 + n;
      const float bb  = bias[col];
#pragma unroll
      for (int i = 0; i < 8; ++i) {  // C layout: VGPR i holds row i + 8*hf
        Out[(size_t)(row0 + hf * 8 + i) * H_ + col]      = (__bf16)(acc0[t][i] + bb);
        Out[(size_t)(row0 + 16 + hf * 8 + i) * H_ + col] = (__bf16)(acc1[t][i] + bb);
      }
    }
  } else {
    const int bidx = row0 / S_;
    const int sloc = row0 % S_;
#pragma unroll
    for (int t = 0; t < 4; ++t) {
      const int   col = col0 + t * 16 + n;
      const float bb  = bias[col];
      v8bf p0, p1;
#pragma unroll
      for (int i = 0; i < 8; ++i) {
        p0[i] = (__bf16)(acc0[t][i] + bb);
        p1[i] = (__bf16)(acc1[t][i] + bb);
      }
      __bf16* dst = Vt + ((size_t)bidx * H_ + col) * S_ + sloc + hf * 8;
      *(v8bf*)(dst)      = p0;
      *(v8bf*)(dst + 16) = p1;
    }
  }
}

// ---------------- scores = Q K^T / sqrt(H), masked (fp32 logits) ----------------

__global__ void __launch_bounds__(128)
attn_scores(const __bf16* __restrict__ Qb, const __bf16* __restrict__ Kb,
            const int* __restrict__ mask, float* __restrict__ Sc) {
  const int b    = blockIdx.z;
  const int lane = threadIdx.x & 31;
  const int wid  = threadIdx.x >> 5;
  const int row0 = blockIdx.x * 64 + (wid & 1) * 32;   // query
  const int col0 = blockIdx.y * 128 + (wid >> 1) * 64; // key
  const __bf16* Q = Qb + (size_t)b * S_ * H_;
  const __bf16* K = Kb + (size_t)b * S_ * H_;  // [key][d] == K-major B operand

  GEMM_LOOP_BF16(Q, H_, K, H_, H_)

  const int   n = lane & 15, hf = lane >> 4;
  const float scale  = 0.03125f;  // 1/sqrt(1024)
  const float neginf = -__builtin_inff();
  float*     Srow = Sc + (size_t)b * S_ * S_;
  const int* mrow = mask + (size_t)b * S_;
#pragma unroll
  for (int t = 0; t < 4; ++t) {
    const int  key = col0 + t * 16 + n;
    const bool ok  = (mrow[key] != 0);
#pragma unroll
    for (int i = 0; i < 8; ++i) {
      Srow[(size_t)(row0 + hf * 8 + i) * S_ + key]      = ok ? acc0[t][i] * scale : neginf;
      Srow[(size_t)(row0 + 16 + hf * 8 + i) * S_ + key] = ok ? acc1[t][i] * scale : neginf;
    }
  }
}

// ---------------- row softmax: fp32 logits in -> bf16 probabilities out -----
// 256 threads (8 wave32) per row of 2048; shuffle + LDS reduction. The WMMA
// A operand is bf16 anyway, so emitting bf16 here loses nothing and halves
// the probability-matrix traffic while removing cvt from the PV inner loop.

__global__ void __launch_bounds__(256)
attn_softmax(const float* __restrict__ Sc, __bf16* __restrict__ Pb) {
  const float* row = Sc + (size_t)blockIdx.x * S_;
  __bf16*      out = Pb + (size_t)blockIdx.x * S_;
  const int tid = threadIdx.x;
  float v[8];
  float m = -__builtin_inff();
#pragma unroll
  for (int i = 0; i < 8; ++i) { v[i] = row[tid + i * 256]; m = fmaxf(m, v[i]); }
#pragma unroll
  for (int off = 16; off; off >>= 1) m = fmaxf(m, __shfl_xor(m, off, 32));
  __shared__ float red[8];
  if ((tid & 31) == 0) red[tid >> 5] = m;
  __syncthreads();
  m = red[0];
#pragma unroll
  for (int w = 1; w < 8; ++w) m = fmaxf(m, red[w]);

  float s = 0.f;
#pragma unroll
  for (int i = 0; i < 8; ++i) { v[i] = __expf(v[i] - m); s += v[i]; }
#pragma unroll
  for (int off = 16; off; off >>= 1) s += __shfl_xor(s, off, 32);
  __syncthreads();
  if ((tid & 31) == 0) red[tid >> 5] = s;
  __syncthreads();
  s = 0.f;
#pragma unroll
  for (int w = 0; w < 8; ++w) s += red[w];
  const float inv = (s > 0.f) ? 1.f / s : 0.f;
#pragma unroll
  for (int i = 0; i < 8; ++i) out[tid + i * 256] = (__bf16)(v[i] * inv);
}

// ---------------- out = P V  (pure bf16 microkernel) ----------------

__global__ void __launch_bounds__(128)
attn_pv(const __bf16* __restrict__ Pb, const __bf16* __restrict__ Vt,
        float* __restrict__ Out) {
  const int b    = blockIdx.z;
  const int lane = threadIdx.x & 31;
  const int wid  = threadIdx.x >> 5;
  const int row0 = blockIdx.x * 64 + (wid & 1) * 32;   // query
  const int col0 = blockIdx.y * 128 + (wid >> 1) * 64; // out d
  const __bf16* P = Pb + (size_t)b * S_ * S_;  // [q][key]
  const __bf16* V = Vt + (size_t)b * H_ * S_;  // [d][key] K-major

  GEMM_LOOP_BF16(P, S_, V, S_, S_)

  const int n = lane & 15, hf = lane >> 4;
  float* O = Out + (size_t)b * S_ * H_;
#pragma unroll
  for (int t = 0; t < 4; ++t) {
    const int d = col0 + t * 16 + n;
#pragma unroll
    for (int i = 0; i < 8; ++i) {
      O[(size_t)(row0 + hf * 8 + i) * H_ + d]      = acc0[t][i];
      O[(size_t)(row0 + 16 + hf * 8 + i) * H_ + d] = acc1[t][i];
    }
  }
}

// ---------------- launch ----------------

extern "C" void kernel_launch(void* const* d_in, const int* in_sizes, int n_in,
                              void* d_out, int out_size, void* d_ws, size_t ws_size,
                              hipStream_t stream) {
  (void)in_sizes; (void)n_in; (void)out_size; (void)ws_size;
  const float* X    = (const float*)d_in[0];
  const int*   mask = (const int*)d_in[1];
  const float* Wq   = (const float*)d_in[2];
  const float* bq   = (const float*)d_in[3];
  const float* Wk   = (const float*)d_in[4];
  const float* bk   = (const float*)d_in[5];
  const float* Wv   = (const float*)d_in[6];
  const float* bv   = (const float*)d_in[7];
  float* Out = (float*)d_out;

  char* ws = (char*)d_ws;
  const size_t xb_off = 0;                                   //  32 MB bf16 X
  const size_t wt_off = xb_off + (size_t)BS_ * H_ * 2;       //   6 MB bf16 Wt x3
  const size_t qb_off = wt_off + (size_t)3 * H_ * H_ * 2;    //  32 MB bf16 Q
  const size_t kb_off = qb_off + (size_t)BS_ * H_ * 2;       //  32 MB bf16 K
  const size_t vt_off = kb_off + (size_t)BS_ * H_ * 2;       //  32 MB bf16 V^T
  const size_t sc_off = vt_off + (size_t)BS_ * H_ * 2;       // 128 MB f32 logits
  const size_t pb_off = sc_off + (size_t)B_ * S_ * S_ * 4;   //  64 MB bf16 probs
  __bf16* Xb = (__bf16*)(ws + xb_off);
  __bf16* Wt = (__bf16*)(ws + wt_off);
  __bf16* Qb = (__bf16*)(ws + qb_off);
  __bf16* Kb = (__bf16*)(ws + kb_off);
  __bf16* Vt = (__bf16*)(ws + vt_off);
  float*  Sc = (float*) (ws + sc_off);
  __bf16* Pb = (__bf16*)(ws + pb_off);

  attn_cvt_x<<<(BS_ * H_) / 256, 256, 0, stream>>>(X, Xb);
  attn_cvt_w<<<dim3(H_ / 32, H_ / 32, 3), dim3(32, 8), 0, stream>>>(Wq, Wk, Wv, Wt);
  attn_qkv<<<dim3(BS_ / 64, H_ / 128, 3), 128, 0, stream>>>(Xb, Wt, bq, bk, bv,
                                                            Qb, Kb, Vt);
  attn_scores<<<dim3(S_ / 64, S_ / 128, B_), 128, 0, stream>>>(Qb, Kb, mask, Sc);
  attn_softmax<<<BS_, 256, 0, stream>>>(Sc, Pb);
  attn_pv<<<dim3(S_ / 64, H_ / 128, B_), 128, 0, stream>>>(Pb, Vt, Out);
}
